// TransformerModel_21019569947257
// MI455X (gfx1250) — compile-verified, matrix operand order checked by reading, not run on hardware
//
#include <hip/hip_runtime.h>
#include <math.h>

// ---------------------------------------------------------------------------
// TransformerModel forward for MI455X (gfx1250, wave32, WMMA).
// fp32 I/O, bf16 WMMA compute with f32 accumulate.
// Fragment I/O is organized so every WMMA operand loads as b128 vectors:
// a lane's 16 bf16 fragment elements are two contiguous 8-element K-runs
// (K = hh*8+0..7 and hh*8+16..23), so LDS tiles are laid out K-contiguous
// with 16B-aligned row strides. All transposed staging stores are packed
// bf16x2 (b32) along the K-contiguous axis.
// ---------------------------------------------------------------------------

typedef __attribute__((ext_vector_type(16))) __bf16 v16bf;
typedef __attribute__((ext_vector_type(8)))  float  v8f;

#define D_    1024
#define H_    16
#define HD_   64
#define FF_   4096
#define V_    32000
#define SLEN_ 1024
#define BS_   2
#define NTOK_ (BS_ * SLEN_)
#define EPS_  1e-12f

union F16x8  { uint4 u;    __bf16 e[8];  };
union F16x16 { uint4 u[2]; v16bf  v;     };
union F16x2  { unsigned u; __bf16 e[2];  };

__device__ __forceinline__ v8f wmma_bf16(v16bf a, v16bf b, v8f c) {
    return __builtin_amdgcn_wmma_f32_16x16x32_bf16(false, a, false, b, (short)0, c,
                                                   false, false);
}

// ---------------------------------------------------------------------------
// Embedding gather + LayerNorm + length mask.
// ---------------------------------------------------------------------------
__global__ __launch_bounds__(256)
void k_embed(const int* __restrict__ x, const int* __restrict__ lengths,
             const float* __restrict__ tok_emb, const float* __restrict__ pos_emb,
             const float* __restrict__ g, const float* __restrict__ beta,
             float* __restrict__ h, __bf16* __restrict__ hb)
{
    int row = blockIdx.x;                 // 0..2047
    int b = row / SLEN_, s = row % SLEN_;
    int t = threadIdx.x;
    __shared__ float red[256];

    const float* te = tok_emb + (size_t)x[row] * D_;
    const float* pe = pos_emb + (size_t)s * D_;

    float v[4];
    float lsum = 0.f;
    #pragma unroll
    for (int e = 0; e < 4; ++e) { int c = t * 4 + e; v[e] = te[c] + pe[c]; lsum += v[e]; }
    red[t] = lsum; __syncthreads();
    for (int st = 128; st > 0; st >>= 1) { if (t < st) red[t] += red[t + st]; __syncthreads(); }
    float mean = red[0] * (1.f / D_);
    __syncthreads();
    float lvar = 0.f;
    #pragma unroll
    for (int e = 0; e < 4; ++e) { float d = v[e] - mean; lvar += d * d; }
    red[t] = lvar; __syncthreads();
    for (int st = 128; st > 0; st >>= 1) { if (t < st) red[t] += red[t + st]; __syncthreads(); }
    float inv = rsqrtf(red[0] * (1.f / D_) + EPS_);
    float msk = (s < lengths[b]) ? 1.f : 0.f;

    #pragma unroll
    for (int e = 0; e < 4; ++e) {
        int c = t * 4 + e;
        float o = (g[c] * (v[e] - mean) * inv + beta[c]) * msk;
        h[(size_t)row * D_ + c]  = o;
        hb[(size_t)row * D_ + c] = (__bf16)o;
    }
}

// ---------------------------------------------------------------------------
// Residual add + LayerNorm (+ optional length mask). One block per token row.
// ---------------------------------------------------------------------------
__global__ __launch_bounds__(256)
void k_add_ln(const float* __restrict__ resid, const float* __restrict__ delta,
              const float* __restrict__ g, const float* __restrict__ beta,
              const int* __restrict__ lengths, int apply_mask,
              float* __restrict__ h, __bf16* __restrict__ hb)
{
    int row = blockIdx.x;
    int b = row / SLEN_, s = row % SLEN_;
    int t = threadIdx.x;
    __shared__ float red[256];

    float v[4];
    float lsum = 0.f;
    #pragma unroll
    for (int e = 0; e < 4; ++e) {
        int c = t * 4 + e;
        v[e] = resid[(size_t)row * D_ + c] + delta[(size_t)row * D_ + c];
        lsum += v[e];
    }
    red[t] = lsum; __syncthreads();
    for (int st = 128; st > 0; st >>= 1) { if (t < st) red[t] += red[t + st]; __syncthreads(); }
    float mean = red[0] * (1.f / D_);
    __syncthreads();
    float lvar = 0.f;
    #pragma unroll
    for (int e = 0; e < 4; ++e) { float d = v[e] - mean; lvar += d * d; }
    red[t] = lvar; __syncthreads();
    for (int st = 128; st > 0; st >>= 1) { if (t < st) red[t] += red[t + st]; __syncthreads(); }
    float inv = rsqrtf(red[0] * (1.f / D_) + EPS_);
    float msk = (apply_mask && s >= lengths[b]) ? 0.f : 1.f;

    #pragma unroll
    for (int e = 0; e < 4; ++e) {
        int c = t * 4 + e;
        float o = (g[c] * (v[e] - mean) * inv + beta[c]) * msk;
        h[(size_t)row * D_ + c]  = o;
        hb[(size_t)row * D_ + c] = (__bf16)o;
    }
}

// ---------------------------------------------------------------------------
// GEMM: C[M,N] = A_bf16[M,K] * B_f32[K,N] + bias, optional erf-GELU.
// Block tile 128x64, 8 waves, each wave 32x32 via 2x2 WMMA 16x16x32 bf16.
// LDS row stride = 40 bf16 (80 B, multiple of 16 B) so every fragment is
// exactly two ds_load_b128. B tile is stored transposed (n-major) during the
// f32->bf16 staging pass; each staging thread covers two K-rows x 4 columns
// so conversions pack (v_cvt_pk_bf16_f32) and store as b32.
// ---------------------------------------------------------------------------
#define TM 128
#define TN 64
#define TK 32
#define LPAD 8   // row stride TK+LPAD = 40 elems = 80 B

__global__ __launch_bounds__(256)
void k_gemm(const __bf16* __restrict__ A, const float* __restrict__ B,
            const float* __restrict__ bias, int M, int N, int K,
            int fuse_gelu, float* __restrict__ outF, __bf16* __restrict__ outB)
{
    __shared__ __bf16 As[TM][TK + LPAD];    // m-major, K contiguous
    __shared__ __bf16 Bt[TN][TK + LPAD];    // n-major (transposed), K contiguous

    int t = threadIdx.x;
    int blockM = blockIdx.y * TM, blockN = blockIdx.x * TN;
    int lane = t & 31, w = t >> 5;
    int ln = lane & 15, hh = lane >> 4;
    int wm = (w & 3) * 32, wn = (w >> 2) * 32;

    v8f vzero = {0.f, 0.f, 0.f, 0.f, 0.f, 0.f, 0.f, 0.f};
    v8f acc[2][2];
    acc[0][0] = vzero; acc[0][1] = vzero; acc[1][0] = vzero; acc[1][1] = vzero;

    int arow = t >> 1, ahalf = (t & 1) * 16;     // 128 rows x 2 threads/row
    int bcg  = t & 15, brp = t >> 4;             // 16 col-groups x 16 k-row-pairs

    for (int k0 = 0; k0 < K; k0 += TK) {
        // ---- stage A tile (bf16 activations): 2x b128 load, 2x b128 store ----
        const __bf16* ap = A + (size_t)(blockM + arow) * K + k0 + ahalf;
        uint4 a0 = *reinterpret_cast<const uint4*>(ap);
        uint4 a1 = *reinterpret_cast<const uint4*>(ap + 8);
        *reinterpret_cast<uint4*>(&As[arow][ahalf])     = a0;
        *reinterpret_cast<uint4*>(&As[arow][ahalf + 8]) = a1;

        // ---- stage B tile: two K-rows x 4 cols -> packed bf16x2 b32 stores ----
        const float* bp = B + (size_t)(k0 + 2 * brp) * N + blockN + bcg * 4;
        float4 f0 = *reinterpret_cast<const float4*>(bp);       // row k
        float4 f1 = *reinterpret_cast<const float4*>(bp + N);   // row k+1
        {
            F16x2 pk;
            pk.e[0] = (__bf16)f0.x; pk.e[1] = (__bf16)f1.x;
            *reinterpret_cast<unsigned*>(&Bt[bcg * 4 + 0][2 * brp]) = pk.u;
            pk.e[0] = (__bf16)f0.y; pk.e[1] = (__bf16)f1.y;
            *reinterpret_cast<unsigned*>(&Bt[bcg * 4 + 1][2 * brp]) = pk.u;
            pk.e[0] = (__bf16)f0.z; pk.e[1] = (__bf16)f1.z;
            *reinterpret_cast<unsigned*>(&Bt[bcg * 4 + 2][2 * brp]) = pk.u;
            pk.e[0] = (__bf16)f0.w; pk.e[1] = (__bf16)f1.w;
            *reinterpret_cast<unsigned*>(&Bt[bcg * 4 + 3][2 * brp]) = pk.u;
        }

        if (k0 + TK < K) {   // L2 prefetch of next tiles (global_prefetch_b8)
            __builtin_prefetch(bp + (size_t)TK * N, 0, 1);
            __builtin_prefetch(ap + TK, 0, 1);
        }
        __syncthreads();

        // ---- fragments: two ds_load_b128 each ----
        v16bf af[2], bfr[2];
        #pragma unroll
        for (int i = 0; i < 2; ++i) {
            int m = wm + i * 16 + ln;
            F16x16 fa;
            fa.u[0] = *reinterpret_cast<const uint4*>(&As[m][hh * 8]);
            fa.u[1] = *reinterpret_cast<const uint4*>(&As[m][hh * 8 + 16]);
            af[i] = fa.v;
        }
        #pragma unroll
        for (int j = 0; j < 2; ++j) {
            int n = wn + j * 16 + ln;
            F16x16 fb;
            fb.u[0] = *reinterpret_cast<const uint4*>(&Bt[n][hh * 8]);
            fb.u[1] = *reinterpret_cast<const uint4*>(&Bt[n][hh * 8 + 16]);
            bfr[j] = fb.v;
        }
        #pragma unroll
        for (int i = 0; i < 2; ++i)
            #pragma unroll
            for (int j = 0; j < 2; ++j)
                acc[i][j] = wmma_bf16(af[i], bfr[j], acc[i][j]);
        __syncthreads();
    }

    // ---- epilogue: bias (+ GELU), write f32 and/or bf16 ----
    #pragma unroll
    for (int i = 0; i < 2; ++i) {
        #pragma unroll
        for (int j = 0; j < 2; ++j) {
            #pragma unroll
            for (int r = 0; r < 8; ++r) {
                int grow = blockM + wm + i * 16 + r + 8 * hh;
                int gcol = blockN + wn + j * 16 + ln;
                float val = acc[i][j][r] + bias[gcol];
                if (fuse_gelu) val = 0.5f * val * (1.f + erff(val * 0.70710678118f));
                if (outF) outF[(size_t)grow * N + gcol] = val;
                if (outB) outB[(size_t)grow * N + gcol] = (__bf16)val;
            }
        }
    }
}

// ---------------------------------------------------------------------------
// Causal flash attention: one wave per (batch, head, 16-query tile).
// QK^T and P*V via WMMA bf16; online softmax with 16-lane shfl butterflies.
// Q/K fragments: two global_load_b128 each. V tile staged transposed in LDS
// (packed bf16x2 b32 stores) so each P*V B-fragment is one ds_load_b128
// (upper K half zero-padded).
// ---------------------------------------------------------------------------
__global__ __launch_bounds__(32)
void k_attn(const __bf16* __restrict__ q, const __bf16* __restrict__ k,
            const __bf16* __restrict__ v, __bf16* __restrict__ ctx)
{
    int bid  = blockIdx.x;
    int qt   = bid & 63;            // 64 query tiles of 16
    int head = (bid >> 6) & 15;
    int b    = bid >> 10;
    int lane = threadIdx.x;
    int ln = lane & 15, hh = lane >> 4;

    __shared__ __bf16 p_lds[16][24];   // [q row][key], 48 B stride
    __shared__ __bf16 vt[HD_][24];     // [head dim][key], transposed V tile

    const size_t base = (size_t)b * SLEN_ * D_ + head * HD_;

    // Q fragments (head dim K = 0..31 / 32..63); lane m = ln.
    v16bf qa[2];
    const __bf16* qptr = q + base + (size_t)(qt * 16 + ln) * D_;
    #pragma unroll
    for (int f = 0; f < 2; ++f) {
        F16x16 fq;
        fq.u[0] = *reinterpret_cast<const uint4*>(qptr + f * 32 + hh * 8);
        fq.u[1] = *reinterpret_cast<const uint4*>(qptr + f * 32 + hh * 8 + 16);
        qa[f] = fq.v;
    }

    v8f vzero = {0.f, 0.f, 0.f, 0.f, 0.f, 0.f, 0.f, 0.f};
    float m_run[8], l_run[8];
    v8f o_acc[4];
    #pragma unroll
    for (int r = 0; r < 8; ++r) { m_run[r] = -INFINITY; l_run[r] = 0.f; }
    #pragma unroll
    for (int j = 0; j < 4; ++j) o_acc[j] = vzero;

    const float scale = 0.125f;   // 1/sqrt(64)

    for (int kt = 0; kt <= qt; ++kt) {
        // K^T B-fragments: lane n = key row kt*16+ln, K index = head dim.
        v16bf kf[2];
        const __bf16* kptr = k + base + (size_t)(kt * 16 + ln) * D_;
        #pragma unroll
        for (int f = 0; f < 2; ++f) {
            F16x16 fk;
            fk.u[0] = *reinterpret_cast<const uint4*>(kptr + f * 32 + hh * 8);
            fk.u[1] = *reinterpret_cast<const uint4*>(kptr + f * 32 + hh * 8 + 16);
            kf[f] = fk.v;
        }
        v8f s = vzero;
        s = wmma_bf16(qa[0], kf[0], s);
        s = wmma_bf16(qa[1], kf[1], s);

        // stage V tile (16 keys x 64 dims) transposed: vt[dim][key].
        // Lane covers key-pair (2kp, 2kp+1) x 16 dims -> packed b32 stores.
        {
            int kp = lane & 7;          // key pair index
            int dg = lane >> 3;         // dim group, 16 dims each
            const __bf16* vr = v + base + (size_t)(kt * 16 + 2 * kp) * D_ + dg * 16;
            F16x8 a0, a1, b0, b1;
            a0.u = *reinterpret_cast<const uint4*>(vr);
            a1.u = *reinterpret_cast<const uint4*>(vr + 8);
            b0.u = *reinterpret_cast<const uint4*>(vr + D_);
            b1.u = *reinterpret_cast<const uint4*>(vr + D_ + 8);
            #pragma unroll
            for (int e = 0; e < 8; ++e) {
                F16x2 pk;
                pk.e[0] = a0.e[e]; pk.e[1] = b0.e[e];
                *reinterpret_cast<unsigned*>(&vt[dg * 16 + e][2 * kp]) = pk.u;
                pk.e[0] = a1.e[e]; pk.e[1] = b1.e[e];
                *reinterpret_cast<unsigned*>(&vt[dg * 16 + 8 + e][2 * kp]) = pk.u;
            }
        }

        // scale + causal mask + online softmax update (row = r + 8*hh)
        #pragma unroll
        for (int r = 0; r < 8; ++r) {
            float sv = s[r] * scale;
            int qpos = qt * 16 + r + 8 * hh;
            int kpos = kt * 16 + ln;
            if (kpos > qpos) sv = -INFINITY;
            float mx = sv;
            mx = fmaxf(mx, __shfl_xor(mx, 1, 32));
            mx = fmaxf(mx, __shfl_xor(mx, 2, 32));
            mx = fmaxf(mx, __shfl_xor(mx, 4, 32));
            mx = fmaxf(mx, __shfl_xor(mx, 8, 32));
            float mn = fmaxf(m_run[r], mx);
            float pv = expf(sv - mn);
            float fr = expf(m_run[r] - mn);
            float ps = pv;
            ps += __shfl_xor(ps, 1, 32);
            ps += __shfl_xor(ps, 2, 32);
            ps += __shfl_xor(ps, 4, 32);
            ps += __shfl_xor(ps, 8, 32);
            l_run[r] = l_run[r] * fr + ps;
            m_run[r] = mn;
            #pragma unroll
            for (int j = 0; j < 4; ++j) o_acc[j][r] = o_acc[j][r] * fr;
            p_lds[r + 8 * hh][ln] = (__bf16)pv;
        }
        __syncthreads();

        // P as A-fragment: one ds_load_b128, K = 16 keys zero-padded to 32.
        F16x16 fp;
        fp.u[0] = *reinterpret_cast<const uint4*>(&p_lds[ln][hh * 8]);
        fp.u[1] = make_uint4(0u, 0u, 0u, 0u);
        v16bf pa = fp.v;

        // O += P * V over 4 N-tiles of the 64-wide head dim.
        #pragma unroll
        for (int j = 0; j < 4; ++j) {
            F16x16 fv;
            fv.u[0] = *reinterpret_cast<const uint4*>(&vt[j * 16 + ln][hh * 8]);
            fv.u[1] = make_uint4(0u, 0u, 0u, 0u);
            o_acc[j] = wmma_bf16(pa, fv.v, o_acc[j]);
        }
        __syncthreads();
    }

    // normalize + write bf16 context
    #pragma unroll
    for (int j = 0; j < 4; ++j)
        #pragma unroll
        for (int r = 0; r < 8; ++r) {
            float o = o_acc[j][r] / l_run[r];
            int tok = qt * 16 + r + 8 * hh;
            ctx[((size_t)b * SLEN_ + tok) * D_ + head * HD_ + j * 16 + ln] = (__bf16)o;
        }
}

// ---------------------------------------------------------------------------
// Host-side orchestration.
// ---------------------------------------------------------------------------
extern "C" void kernel_launch(void* const* d_in, const int* in_sizes, int n_in,
                              void* d_out, int out_size, void* d_ws, size_t ws_size,
                              hipStream_t stream) {
    (void)in_sizes; (void)n_in; (void)out_size; (void)ws_size;

    const int*   x        = (const int*)d_in[0];
    const int*   lengths  = (const int*)d_in[1];
    const float* tok_emb  = (const float*)d_in[2];
    const float* pos_emb  = (const float*)d_in[3];
    const float* ln_emb_g = (const float*)d_in[4];
    const float* ln_emb_b = (const float*)d_in[5];
    const float* Wq = (const float*)d_in[6];  const float* bq = (const float*)d_in[7];
    const float* Wk = (const float*)d_in[8];  const float* bk = (const float*)d_in[9];
    const float* Wv = (const float*)d_in[10]; const float* bv = (const float*)d_in[11];
    const float* Wo = (const float*)d_in[12]; const float* bo = (const float*)d_in[13];
    const float* ln1_g = (const float*)d_in[14]; const float* ln1_b = (const float*)d_in[15];
    const float* W1 = (const float*)d_in[16]; const float* b1 = (const float*)d_in[17];
    const float* W2 = (const float*)d_in[18]; const float* b2 = (const float*)d_in[19];
    const float* ln2_g = (const float*)d_in[20]; const float* ln2_b = (const float*)d_in[21];
    const float* projW = (const float*)d_in[22]; const float* projb = (const float*)d_in[23];
    float* out = (float*)d_out;

    char* ws = (char*)d_ws;
    float*  h   = (float*)(ws);                      //  8 MB f32  [2048,1024]
    float*  tmp = (float*)(ws + 8388608);            //  8 MB f32  [2048,1024]
    __bf16* hb  = (__bf16*)(ws + 16777216);          //  4 MB bf16 [2048,1024]
    __bf16* qb  = (__bf16*)(ws + 20971520);          //  4 MB
    __bf16* kb  = (__bf16*)(ws + 25165824);          //  4 MB
    __bf16* vb  = (__bf16*)(ws + 29360128);          //  4 MB
    __bf16* cb  = (__bf16*)(ws + 33554432);          //  4 MB
    __bf16* ff1 = (__bf16*)(ws + 37748736);          // 16 MB bf16 [2048,4096]

    k_embed<<<NTOK_, 256, 0, stream>>>(x, lengths, tok_emb, pos_emb, ln_emb_g, ln_emb_b,
                                       h, hb);

    for (int i = 0; i < 2; ++i) {
        dim3 gP(D_ / TN, NTOK_ / TM);
        size_t wofs = (size_t)i * D_ * D_;
        k_gemm<<<gP, 256, 0, stream>>>(hb, Wq + wofs, bq + i * D_, NTOK_, D_, D_, 0,
                                       nullptr, qb);
        k_gemm<<<gP, 256, 0, stream>>>(hb, Wk + wofs, bk + i * D_, NTOK_, D_, D_, 0,
                                       nullptr, kb);
        k_gemm<<<gP, 256, 0, stream>>>(hb, Wv + wofs, bv + i * D_, NTOK_, D_, D_, 0,
                                       nullptr, vb);

        k_attn<<<BS_ * H_ * (SLEN_ / 16), 32, 0, stream>>>(qb, kb, vb, cb);

        k_gemm<<<gP, 256, 0, stream>>>(cb, Wo + wofs, bo + i * D_, NTOK_, D_, D_, 0,
                                       tmp, nullptr);
        k_add_ln<<<NTOK_, 256, 0, stream>>>(h, tmp, ln1_g + i * D_, ln1_b + i * D_,
                                            lengths, 0, h, hb);

        dim3 gF1(FF_ / TN, NTOK_ / TM);
        k_gemm<<<gF1, 256, 0, stream>>>(hb, W1 + (size_t)i * D_ * FF_, b1 + i * FF_,
                                        NTOK_, FF_, D_, 1, nullptr, ff1);
        dim3 gF2(D_ / TN, NTOK_ / TM);
        k_gemm<<<gF2, 256, 0, stream>>>(ff1, W2 + (size_t)i * FF_ * D_, b2 + i * D_,
                                        NTOK_, D_, FF_, 0, tmp, nullptr);
        k_add_ln<<<NTOK_, 256, 0, stream>>>(h, tmp, ln2_g + i * D_, ln2_b + i * D_,
                                            lengths, 1, h, hb);
    }

    dim3 gV(V_ / TN, NTOK_ / TM);
    k_gemm<<<gV, 256, 0, stream>>>(hb, projW, projb, NTOK_, V_, D_, 0, out, nullptr);
}
